// EventDrivenSparseNetwork_62869731279040
// MI455X (gfx1250) — matrix operand order repaired and checked by reading, not compiled, not run on hardware
//
#include <hip/hip_runtime.h>

// ---------------------------------------------------------------------------
// Sparse spiking network: 4x [ masked GEMM (bf16 WMMA) -> LIF scan over T ]
// B=32, T=100, N=2048.  X:[B*T,N] @ (W*mask)^T : [N,N] -> cur:[B*T,N]
// ---------------------------------------------------------------------------

#define NN     2048          // feature dim (both in and out)
#define BT     3200          // B*T rows
#define TT     100
#define BB     32
#define KSTEP  32            // K per v_wmma_f32_16x16x32_bf16
#define ALPHA  0.95122942450071400910f   // exp(-1/20)

typedef __bf16 v8bf  __attribute__((ext_vector_type(8)));
typedef __bf16 v16bf __attribute__((ext_vector_type(16)));
typedef float  v8f   __attribute__((ext_vector_type(8)));

static __device__ __forceinline__ v16bf cat16(v8bf lo, v8bf hi) {
    return __builtin_shufflevector(lo, hi, 0,1,2,3,4,5,6,7,8,9,10,11,12,13,14,15);
}

// -------------------------------- f32 -> bf16 ------------------------------
__global__ __launch_bounds__(256)
void cvt_f32_bf16_kernel(const float* __restrict__ src, __bf16* __restrict__ dst, int n) {
    int i = blockIdx.x * 256 + threadIdx.x;
    if (i < n) dst[i] = (__bf16)src[i];
}

// ------------------------- W*mask -> bf16 (fused) --------------------------
__global__ __launch_bounds__(256)
void maskw_kernel(const float* __restrict__ W, const unsigned char* __restrict__ mask,
                  __bf16* __restrict__ Wb, int n) {
    int i = blockIdx.x * 256 + threadIdx.x;
    if (i < n) Wb[i] = (__bf16)(mask[i] ? W[i] : 0.0f);
}

// ------------------------------- WMMA GEMM ---------------------------------
// C[BT,NN] = A[BT,NN(k)] * Wb[NN(n),NN(k)]^T, f32 accum.
// Block: 256 thr = 8 waves; block tile 64(M) x 64(N); wave tile 16(M) x 32(N).
__global__ __launch_bounds__(256)
void gemm_bf16_wmma_kernel(const __bf16* __restrict__ Ag,   // [BT, NN] row-major
                           const __bf16* __restrict__ Bg,   // [NN, NN] row-major (n,k)
                           float* __restrict__ C)           // [BT, NN]
{
    __shared__ __align__(16) __bf16 lA[2][64 * KSTEP];
    __shared__ __align__(16) __bf16 lB[2][64 * KSTEP];

    const int tid  = threadIdx.x;
    const int lane = tid & 31;
    const int wave = tid >> 5;
    const int hi   = lane >> 4;          // 0: lanes 0-15, 1: lanes 16-31
    const int l15  = lane & 15;

    const int m0 = blockIdx.x * 64;      // row base into BT
    const int n0 = blockIdx.y * 64;      // col base into NN
    const int wm = (wave >> 1) * 16;     // wave row offset: 0,16,32,48
    const int wn = (wave & 1)  * 32;     // wave col offset: 0,32

    // Global->LDS staging: 64x32 bf16 tile = 256 chunks of 8 halves (16B each)
    const int ldRow = tid >> 2;          // 0..63
    const int ldCol = (tid & 3) * 8;     // 0,8,16,24

    const __bf16* aSrc = Ag + (size_t)(m0 + ldRow) * NN + ldCol;
    const __bf16* bSrc = Bg + (size_t)(n0 + ldRow) * NN + ldCol;

    v8f acc0 = {};
    v8f acc1 = {};

    // prologue: stage k-tile 0 into buffer 0
    v8bf ra = *(const v8bf*)aSrc;
    v8bf rb = *(const v8bf*)bSrc;
    *(v8bf*)&lA[0][ldRow * KSTEP + ldCol] = ra;
    *(v8bf*)&lB[0][ldRow * KSTEP + ldCol] = rb;
    __syncthreads();

    const int KT = NN / KSTEP;           // 64 k-tiles
    int buf = 0;
    for (int kt = 0; kt < KT; ++kt) {
        if (kt + 1 < KT) {               // prefetch next k-tile into registers
            ra = *(const v8bf*)(aSrc + (size_t)(kt + 1) * KSTEP);
            rb = *(const v8bf*)(bSrc + (size_t)(kt + 1) * KSTEP);
        }

        // A fragment: lane<16 -> M=l15, K = 0..15; lane>=16 -> K = 16..31 (contiguous)
        const __bf16* ap = &lA[buf][(wm + l15) * KSTEP];
        v16bf afr = cat16(*(const v8bf*)(ap + hi * 16),
                          *(const v8bf*)(ap + hi * 16 + 8));

        // B fragments: lane<16 -> N=l15, K=0..15; lane>=16 -> K=16..31 (contiguous)
        const __bf16* bp0 = &lB[buf][(wn + l15) * KSTEP + hi * 16];
        v16bf bfr0 = cat16(*(const v8bf*)bp0, *(const v8bf*)(bp0 + 8));
        const __bf16* bp1 = bp0 + 16 * KSTEP;       // next 16-col tile
        v16bf bfr1 = cat16(*(const v8bf*)bp1, *(const v8bf*)(bp1 + 8));

        acc0 = __builtin_amdgcn_wmma_f32_16x16x32_bf16(false, afr, false, bfr0,
                                                       (short)0, acc0, false, false);
        acc1 = __builtin_amdgcn_wmma_f32_16x16x32_bf16(false, afr, false, bfr1,
                                                       (short)0, acc1, false, false);

        if (kt + 1 < KT) {               // write prefetched tile into other buffer
            *(v8bf*)&lA[buf ^ 1][ldRow * KSTEP + ldCol] = ra;
            *(v8bf*)&lB[buf ^ 1][ldRow * KSTEP + ldCol] = rb;
        }
        __syncthreads();
        buf ^= 1;
    }

    // Epilogue: C/D layout — lane<16: VGPR j -> M=j, N=l15; lane>=16: M=8+j
    float* cp = C + (size_t)(m0 + wm + hi * 8) * NN + (n0 + wn + l15);
    #pragma unroll
    for (int j = 0; j < 8; ++j) {
        cp[(size_t)j * NN]      = acc0[j];
        cp[(size_t)j * NN + 16] = acc1[j];
    }
}

// ------------------------------- LIF scan ----------------------------------
// One thread per (b, m); v kept in a register across the T=100 recurrence.
__global__ __launch_bounds__(256)
void lif_kernel(const float* __restrict__ cur, __bf16* __restrict__ spikes,
                float* __restrict__ outF) {
    int idx = blockIdx.x * 256 + threadIdx.x;   // 0 .. B*NN-1
    int b = idx >> 11;                          // / NN
    int m = idx & (NN - 1);
    size_t base = (size_t)b * TT * NN + m;

    float v = 0.0f;
    #pragma unroll 4
    for (int t = 0; t < TT; ++t) {
        float I = cur[base + (size_t)t * NN];
        v = ALPHA * v + I;
        float s = (v >= 1.0f) ? 1.0f : 0.0f;
        v = (v >= 1.0f) ? 0.0f : v;
        spikes[base + (size_t)t * NN] = (__bf16)s;
        if (outF) outF[base + (size_t)t * NN] = s;
    }
}

// ------------------------------- launcher ----------------------------------
extern "C" void kernel_launch(void* const* d_in, const int* in_sizes, int n_in,
                              void* d_out, int out_size, void* d_ws, size_t ws_size,
                              hipStream_t stream) {
    // inputs (setup_inputs dict order): x, W0, mask0, W1, mask1, W2, mask2, W3, mask3
    const float* x = (const float*)d_in[0];
    const float* Ws[4]         = { (const float*)d_in[1], (const float*)d_in[3],
                                   (const float*)d_in[5], (const float*)d_in[7] };
    const unsigned char* Ms[4] = { (const unsigned char*)d_in[2], (const unsigned char*)d_in[4],
                                   (const unsigned char*)d_in[6], (const unsigned char*)d_in[8] };

    const size_t nAct = (size_t)BT * NN;        // 6,553,600
    const size_t nW   = (size_t)NN * NN;        // 4,194,304

    // workspace layout (all offsets 256B-aligned)
    char* w = (char*)d_ws;
    __bf16* actA = (__bf16*)(w);                          // 13,107,200 B
    __bf16* actB = (__bf16*)(w + nAct * 2);               // 13,107,200 B
    float*  cur  = (float*) (w + nAct * 4);               // 26,214,400 B
    __bf16* Wb   = (__bf16*)(w + nAct * 4 + nAct * 4);    //  8,388,608 B  (~61 MB total)

    // layer-0 input -> bf16
    cvt_f32_bf16_kernel<<<(int)((nAct + 255) / 256), 256, 0, stream>>>(x, actA, (int)nAct);

    __bf16* aIn  = actA;
    __bf16* aOut = actB;
    dim3 ggrid(BT / 64, NN / 64);               // 50 x 32
    for (int l = 0; l < 4; ++l) {
        maskw_kernel<<<(int)((nW + 255) / 256), 256, 0, stream>>>(Ws[l], Ms[l], Wb, (int)nW);
        gemm_bf16_wmma_kernel<<<ggrid, 256, 0, stream>>>(aIn, Wb, cur);
        lif_kernel<<<(BB * NN) / 256, 256, 0, stream>>>(
            cur, aOut, (l == 3) ? (float*)d_out : nullptr);
        __bf16* t = aIn; aIn = aOut; aOut = t;
    }
}
